// MixtureOfExperts_85289460564321
// MI455X (gfx1250) — matrix operand order repaired and checked by reading, not compiled, not run on hardware
//
#include <hip/hip_runtime.h>
#include <hip/hip_bf16.h>
#include <math.h>

#define NEXP 8
#define BB 4
#define SS 2048
#define DD 2048
#define FFDIM 1024
#define NTOK (BB*SS)            // 8192 tokens
#define ROUTER_BLOCKS (NTOK/8)  // 1024 (8 waves/block, 1 token/wave)
#define MROWS 64                // token rows per FFN workgroup
#define TILES64 (NTOK/MROWS)    // 128 worst-case row-tiles per expert

typedef __attribute__((ext_vector_type(16))) __bf16 v16bf;
typedef __attribute__((ext_vector_type(8)))  __bf16 v8bf;
typedef __attribute__((ext_vector_type(8)))  float  v8f;

// ---------------------------------------------------------------------------
// Kernel 1: router. One wave32 per token.
// ---------------------------------------------------------------------------
__global__ __launch_bounds__(256) void moe_router(
    const float* __restrict__ x, const float* __restrict__ Wg,
    int* __restrict__ counts, int* __restrict__ top1,
    float* __restrict__ partial, int* __restrict__ toklist,
    float* __restrict__ wlist)
{
  __shared__ float s_probs[8][NEXP];
  const int wave = threadIdx.x >> 5;
  const int lane = threadIdx.x & 31;
  const int t = blockIdx.x * 8 + wave;

  float acc[NEXP];
#pragma unroll
  for (int e = 0; e < NEXP; ++e) acc[e] = 0.f;

  const float* xr = x + (size_t)t * DD;
  for (int d = lane; d < DD; d += 32) {
    const float xv = xr[d];
    const float* wr = Wg + (size_t)d * NEXP;
#pragma unroll
    for (int e = 0; e < NEXP; ++e) acc[e] = fmaf(xv, wr[e], acc[e]);
  }
#pragma unroll
  for (int off = 16; off >= 1; off >>= 1) {
#pragma unroll
    for (int e = 0; e < NEXP; ++e) acc[e] += __shfl_xor(acc[e], off, 32);
  }

  if (lane == 0) {
    int e0 = 0; float v0 = acc[0];
    for (int e = 1; e < NEXP; ++e) if (acc[e] > v0) { v0 = acc[e]; e0 = e; }
    int e1 = -1; float v1 = -3.0e38f;
    for (int e = 0; e < NEXP; ++e)
      if (e != e0 && acc[e] > v1) { v1 = acc[e]; e1 = e; }

    const float r  = expf(v1 - v0);        // softmax over top-2 (v0 is max)
    const float w0 = 1.f / (1.f + r);
    const float w1 = 1.f - w0;

    int p0 = atomicAdd(&counts[e0], 1);
    toklist[e0 * NTOK + p0] = t; wlist[e0 * NTOK + p0] = w0;
    int p1 = atomicAdd(&counts[e1], 1);
    toklist[e1 * NTOK + p1] = t; wlist[e1 * NTOK + p1] = w1;
    atomicAdd(&top1[e0], 1);

    float s = 0.f, pr[NEXP];
    for (int e = 0; e < NEXP; ++e) { pr[e] = expf(acc[e] - v0); s += pr[e]; }
    const float inv = 1.f / s;
    for (int e = 0; e < NEXP; ++e) s_probs[wave][e] = pr[e] * inv;
  }
  __syncthreads();
  if (threadIdx.x < NEXP) {   // fixed-order per-block reduction (deterministic)
    float s = 0.f;
#pragma unroll
    for (int w = 0; w < 8; ++w) s += s_probs[w][threadIdx.x];
    partial[blockIdx.x * NEXP + threadIdx.x] = s;
  }
}

// ---------------------------------------------------------------------------
// Kernel 2: finalize load-balancing loss, fixed order => deterministic.
// ---------------------------------------------------------------------------
__global__ void moe_loss(const float* __restrict__ partial,
                         const int* __restrict__ top1,
                         float* __restrict__ out_loss)
{
  if (threadIdx.x == 0 && blockIdx.x == 0) {
    float avg[NEXP];
    for (int e = 0; e < NEXP; ++e) avg[e] = 0.f;
    for (int b = 0; b < ROUTER_BLOCKS; ++b)
      for (int e = 0; e < NEXP; ++e) avg[e] += partial[b * NEXP + e];
    float lb = 0.f;
    for (int e = 0; e < NEXP; ++e) {
      const float a = avg[e] / (float)NTOK;           // avg_probs
      const float f = (float)top1[e] / (float)NTOK;   // top-1 frequency
      lb += f * a;
    }
    *out_loss = 0.01f * ((float)NEXP * lb);
  }
}

// ---------------------------------------------------------------------------
// Kernel 3: pack a weight matrix (E, K, N) f32 into WMMA B-fragment-linear
// bf16: frag = (e*(N/16)+ntg)*(K/32)+kci ; per-lane 16 contiguous halfs so the
// FFN kernel loads each fragment with one coalesced 32B load per lane.
// B layout used here (contiguous k, same order as A): lanes 0-15 hold K=kci*32+0..15 of col ntg*16+n,
// lanes 16-31 hold K=kci*32+16..31.
// ---------------------------------------------------------------------------
__global__ __launch_bounds__(256) void pack_w(
    const float* __restrict__ src, __bf16* __restrict__ dst, int K, int N)
{
  const int frag = blockIdx.x * 8 + (threadIdx.x >> 5);
  const int lane = threadIdx.x & 31;
  const int kfr  = K >> 5;                  // K/32
  const int kci  = frag % kfr;
  const int rest = frag / kfr;
  const int ntg  = rest % (N >> 4);
  const int e    = rest / (N >> 4);
  const int col  = ntg * 16 + (lane & 15);
  const int k0   = kci * 32 + (lane >> 4) * 16;
  const float* sp = src + ((size_t)e * K + k0) * N + col;
  v16bf v;
#pragma unroll
  for (int j = 0; j < 16; ++j) v[j] = (__bf16)sp[(size_t)j * N];
  *reinterpret_cast<v16bf*>(dst + (size_t)frag * 512 + lane * 16) = v;
}

// ---------------------------------------------------------------------------
// Kernel 4: fused expert FFN, bf16 WMMA, f32 accumulate.
// One 256-thread workgroup = (expert e, tile of 64 gathered tokens).
// Wave w: row-slab (w&3)*16.. , column half (w>>2).
// Stage 1: H(64x1024) = relu(Xg*W1+b1) -> 128KB LDS (bf16).
// Stage 2: Y = H*W2+b2, scaled by routing weight, atomicAdd into out.
// PRE=true: B fragments from packed bf16 (1 coalesced 32B load/lane).
// PRE=false: fallback, strided f32 gather + cvt.
// ---------------------------------------------------------------------------
template <bool PRE>
__global__ __launch_bounds__(256) void moe_ffn(
    const float* __restrict__ x,
    const float* __restrict__ W1, const __bf16* __restrict__ W1p,
    const float* __restrict__ b1,
    const float* __restrict__ W2, const __bf16* __restrict__ W2p,
    const float* __restrict__ b2,
    const int* __restrict__ counts, const int* __restrict__ toklist,
    const float* __restrict__ wlist, float* __restrict__ out)
{
  const int e   = blockIdx.x >> 7;               // TILES64 = 128
  const int rt  = blockIdx.x & (TILES64 - 1);
  const int cnt = counts[e];
  if (rt * MROWS >= cnt) return;                 // block-uniform exit

  __shared__ int    s_tok[MROWS];
  __shared__ float  s_wt[MROWS];
  __shared__ __bf16 s_h[MROWS * FFDIM];          // 128 KB H tile

  const int tid   = threadIdx.x;
  const int wave  = tid >> 5;
  const int lane  = tid & 31;
  const int row   = lane & 15;
  const int hi    = lane >> 4;
  const int rslab = wave & 3;                    // 4 row slabs of 16
  const int chalf = wave >> 2;                   // 2 column halves

  if (tid < MROWS) {
    const int slot = rt * MROWS + tid;
    if (slot < cnt) { s_tok[tid] = toklist[e * NTOK + slot];
                      s_wt[tid]  = wlist[e * NTOK + slot]; }
    else            { s_tok[tid] = -1; s_wt[tid] = 0.f; }
  }
  __syncthreads();

  const int myrow = rslab * 16 + row;
  const int tokA  = s_tok[myrow];
  const float* xrow = x + (size_t)(tokA < 0 ? 0 : tokA) * DD;
  const float* W1e  = W1 + (size_t)e * DD * FFDIM;
  const float* W2e  = W2 + (size_t)e * FFDIM * DD;

  const v8f vzero = {0.f,0.f,0.f,0.f,0.f,0.f,0.f,0.f};

  // ---------------- Stage 1: H = relu(X * W1 + b1) ----------------
  for (int pass = 0; pass < 2; ++pass) {
    const int n0 = chalf * 512 + pass * 256;
    v8f acc[16];
#pragma unroll
    for (int i = 0; i < 16; ++i) acc[i] = vzero;

    for (int kc = 0; kc < DD; kc += 32) {
      v16bf a;
      {
        const float* ap = xrow + kc + hi * 16;
        const float4 f0 = *reinterpret_cast<const float4*>(ap);
        const float4 f1 = *reinterpret_cast<const float4*>(ap + 4);
        const float4 f2 = *reinterpret_cast<const float4*>(ap + 8);
        const float4 f3 = *reinterpret_cast<const float4*>(ap + 12);
        a[0]=(__bf16)f0.x;  a[1]=(__bf16)f0.y;  a[2]=(__bf16)f0.z;  a[3]=(__bf16)f0.w;
        a[4]=(__bf16)f1.x;  a[5]=(__bf16)f1.y;  a[6]=(__bf16)f1.z;  a[7]=(__bf16)f1.w;
        a[8]=(__bf16)f2.x;  a[9]=(__bf16)f2.y;  a[10]=(__bf16)f2.z; a[11]=(__bf16)f2.w;
        a[12]=(__bf16)f3.x; a[13]=(__bf16)f3.y; a[14]=(__bf16)f3.z; a[15]=(__bf16)f3.w;
      }
#pragma unroll
      for (int nt = 0; nt < 16; ++nt) {
        v16bf b;
        if (PRE) {
          const int ntg = chalf * 32 + pass * 16 + nt;
          const __bf16* bp = W1p +
              ((((size_t)e * 64 + ntg) * 64) + (kc >> 5)) * 512 + lane * 16;
          b = *reinterpret_cast<const v16bf*>(bp);
        } else {
          const int col = n0 + nt * 16 + row;
          const float* bp = W1e + (size_t)(kc + hi * 16) * FFDIM + col;
          float t[16];
#pragma unroll
          for (int j = 0; j < 16; ++j) t[j] = bp[(size_t)j * FFDIM];
#pragma unroll
          for (int j = 0; j < 16; ++j) b[j] = (__bf16)t[j];
        }
        acc[nt] = __builtin_amdgcn_wmma_f32_16x16x32_bf16(
            false, a, false, b, (short)0, acc[nt], false, false);
      }
    }
    // bias + relu -> LDS (C layout: M = v + 8*hi within slab, N = col)
#pragma unroll
    for (int nt = 0; nt < 16; ++nt) {
      const int col  = n0 + nt * 16 + row;
      const float bv = b1[(size_t)e * FFDIM + col];
#pragma unroll
      for (int v = 0; v < 8; ++v) {
        float h = acc[nt][v] + bv;
        h = fmaxf(h, 0.f);
        s_h[(size_t)(rslab * 16 + v + 8 * hi) * FFDIM + col] = (__bf16)h;
      }
    }
  }
  __syncthreads();

  int tokr[8]; float wtr[8];
#pragma unroll
  for (int v = 0; v < 8; ++v) {
    tokr[v] = s_tok[rslab * 16 + v + 8 * hi];
    wtr[v]  = s_wt [rslab * 16 + v + 8 * hi];
  }

  // ---------------- Stage 2: Y = H * W2 + b2, scatter-accumulate ----------
  for (int pass = 0; pass < 4; ++pass) {
    const int m0 = chalf * 1024 + pass * 256;
    v8f acc2[16];
#pragma unroll
    for (int i = 0; i < 16; ++i) acc2[i] = vzero;

    for (int kc = 0; kc < FFDIM; kc += 32) {
      const __bf16* hp = s_h + (size_t)myrow * FFDIM + kc + hi * 16;
      const v8bf lo8 = *reinterpret_cast<const v8bf*>(hp);
      const v8bf hh8 = *reinterpret_cast<const v8bf*>(hp + 8);
      const v16bf a2 = __builtin_shufflevector(
          lo8, hh8, 0,1,2,3,4,5,6,7,8,9,10,11,12,13,14,15);
#pragma unroll
      for (int nt = 0; nt < 16; ++nt) {
        v16bf b;
        if (PRE) {
          const int ntg = chalf * 64 + pass * 16 + nt;
          const __bf16* bp = W2p +
              ((((size_t)e * 128 + ntg) * 32) + (kc >> 5)) * 512 + lane * 16;
          b = *reinterpret_cast<const v16bf*>(bp);
        } else {
          const int col = m0 + nt * 16 + row;
          const float* bp = W2e + (size_t)(kc + hi * 16) * DD + col;
          float t[16];
#pragma unroll
          for (int j = 0; j < 16; ++j) t[j] = bp[(size_t)j * DD];
#pragma unroll
          for (int j = 0; j < 16; ++j) b[j] = (__bf16)t[j];
        }
        acc2[nt] = __builtin_amdgcn_wmma_f32_16x16x32_bf16(
            false, a2, false, b, (short)0, acc2[nt], false, false);
      }
    }
#pragma unroll
    for (int nt = 0; nt < 16; ++nt) {
      const int col  = m0 + nt * 16 + row;
      const float bv = b2[(size_t)e * DD + col];
#pragma unroll
      for (int v = 0; v < 8; ++v) {
        if (tokr[v] >= 0) {
          const float val = (acc2[nt][v] + bv) * wtr[v];
          atomicAdd(&out[(size_t)tokr[v] * DD + col], val);
        }
      }
    }
  }
}

// ---------------------------------------------------------------------------
extern "C" void kernel_launch(void* const* d_in, const int* in_sizes, int n_in,
                              void* d_out, int out_size, void* d_ws, size_t ws_size,
                              hipStream_t stream)
{
  const float* x  = (const float*)d_in[0];
  const float* Wg = (const float*)d_in[1];
  const float* W1 = (const float*)d_in[2];
  const float* b1 = (const float*)d_in[3];
  const float* W2 = (const float*)d_in[4];
  const float* b2 = (const float*)d_in[5];
  float* out = (float*)d_out;

  // Workspace layout:
  const size_t OFF_TOP1    = 32;
  const size_t OFF_PARTIAL = 64;
  const size_t OFF_TOK     = OFF_PARTIAL + (size_t)ROUTER_BLOCKS * NEXP * 4;
  const size_t OFF_WT      = OFF_TOK + (size_t)NEXP * NTOK * 4;
  const size_t OFF_W1P     = OFF_WT + (size_t)NEXP * NTOK * 4;   // 64B aligned
  const size_t NW          = (size_t)NEXP * DD * FFDIM;          // 16.7M elems
  const size_t OFF_W2P     = OFF_W1P + NW * 2;
  const size_t WS_NEEDED   = OFF_W2P + NW * 2;                   // ~67.7 MB

  char* ws = (char*)d_ws;
  int*    counts  = (int*)(ws + 0);
  int*    top1    = (int*)(ws + OFF_TOP1);
  float*  partial = (float*)(ws + OFF_PARTIAL);
  int*    toklist = (int*)(ws + OFF_TOK);
  float*  wlist   = (float*)(ws + OFF_WT);
  __bf16* W1p     = (__bf16*)(ws + OFF_W1P);
  __bf16* W2p     = (__bf16*)(ws + OFF_W2P);

  hipMemsetAsync(d_out, 0, (size_t)out_size * sizeof(float), stream);
  hipMemsetAsync(ws, 0, 64, stream);

  moe_router<<<ROUTER_BLOCKS, 256, 0, stream>>>(x, Wg, counts, top1, partial,
                                                toklist, wlist);
  moe_loss<<<1, 32, 0, stream>>>(partial, top1, out + (size_t)NTOK * DD);

  const dim3 ffn_grid(NEXP * TILES64);
  if (ws_size >= WS_NEEDED) {
    // W1: 8*(1024/16)*(2048/32) = 32768 frags; W2: 8*(2048/16)*(1024/32) same.
    pack_w<<<32768 / 8, 256, 0, stream>>>(W1, W1p, DD, FFDIM);
    pack_w<<<32768 / 8, 256, 0, stream>>>(W2, W2p, FFDIM, DD);
    moe_ffn<true><<<ffn_grid, 256, 0, stream>>>(x, W1, W1p, b1, W2, W2p, b2,
                                                counts, toklist, wlist, out);
  } else {
    moe_ffn<false><<<ffn_grid, 256, 0, stream>>>(x, W1, W1p, b1, W2, W2p, b2,
                                                 counts, toklist, wlist, out);
  }
}
